// ProposalPointGuidedFusion_62878321214197
// MI455X (gfx1250) — matrix operand order repaired and checked by reading, not compiled, not run on hardware
//
#include <hip/hip_runtime.h>
#include <hip/hip_bf16.h>
#include <math.h>

#define P_BOX   512
#define N_PTS   50000
#define K_MAX   64
#define HID     256
#define SAMP    4
#define C_IMG   96
#define HF      160
#define WF      512
#define CAP     4096
#define M_TOT   (P_BOX * K_MAX)   // 32768
#define IMG_W_F 1280.0f
#define IMG_H_F 384.0f

typedef _Float16 v16h __attribute__((ext_vector_type(16)));
typedef float    v8f  __attribute__((ext_vector_type(8)));

// ---------------------------------------------------------------------------
// CDNA5 async global->LDS copy (ASYNCcnt-tracked). VDST operand is the
// wave-relative LDS byte offset = low 32 bits of the generic LDS address.
// ---------------------------------------------------------------------------
__device__ __forceinline__ void async_copy_b128(unsigned lds_off,
                                                const void* gaddr) {
  asm volatile("global_load_async_to_lds_b128 %0, %1, off"
               :: "v"(lds_off), "v"((unsigned long long)gaddr)
               : "memory");
}
__device__ __forceinline__ void wait_async_le2() {
  asm volatile("s_wait_asynccnt 0x2" ::: "memory");
}
__device__ __forceinline__ void wait_async_0() {
  asm volatile("s_wait_asynccnt 0x0" ::: "memory");
}

// ---------------------------------------------------------------------------
// K1: per-box candidate compaction. key = (planar_bits << 32) | point_idx.
// ---------------------------------------------------------------------------
__launch_bounds__(256)
__global__ void k_compact(const float* __restrict__ pts,
                          const float* __restrict__ boxes,
                          unsigned long long* __restrict__ cand,
                          int* __restrict__ counts,
                          int* __restrict__ candcnt) {
  int b = blockIdx.x;
  __shared__ int s_cc;
  if (threadIdx.x == 0) s_cc = 0;
  __syncthreads();
  float cx = boxes[b*7+0], cy = boxes[b*7+1], cz = boxes[b*7+2];
  float dx = boxes[b*7+3], dy = boxes[b*7+4], dz = boxes[b*7+5];
  float yaw = boxes[b*7+6];
  float c = cosf(yaw), s = sinf(yaw);
  float hx = dx*0.5f, hy = dy*0.5f, hz = dz*0.5f;
  float ddx = fmaxf(dx, 0.001f), ddy = fmaxf(dy, 0.001f);
  for (int i = threadIdx.x; i < N_PTS; i += 256) {
    float px = pts[i*7+0], py = pts[i*7+1], pz = pts[i*7+2];
    float rx = px-cx, ry = py-cy, rz = pz-cz;
    float lx =  rx*c + ry*s;
    float ly = -rx*s + ry*c;
    if (fabsf(lx) <= hx && fabsf(ly) <= hy && fabsf(rz) <= hz) {
      float nx = lx/ddx, ny = ly/ddy;
      float planar = sqrtf(nx*nx + ny*ny);
      int slot = atomicAdd(&s_cc, 1);
      if (slot < CAP) {
        unsigned long long key =
            ((unsigned long long)__float_as_uint(planar) << 32) | (unsigned)i;
        cand[(size_t)b*CAP + slot] = key;
      }
    }
  }
  __syncthreads();
  if (threadIdx.x == 0) {
    counts[b]  = s_cc;
    candcnt[b] = (s_cc < CAP) ? s_cc : CAP;
  }
}

// ---------------------------------------------------------------------------
// K2: extract 64 smallest keys per box (membership-exact; order irrelevant
// because the final pooling is permutation-invariant over slots).
// ---------------------------------------------------------------------------
__launch_bounds__(256)
__global__ void k_select(const unsigned long long* __restrict__ cand,
                         const int* __restrict__ counts,
                         const int* __restrict__ candcnt,
                         int* __restrict__ sel, int* __restrict__ nsel) {
  __shared__ unsigned long long keys[CAP];
  __shared__ unsigned long long red[256];
  int b = blockIdx.x, tid = threadIdx.x;
  int cc = candcnt[b];
  for (int i = tid; i < cc; i += 256) keys[i] = cand[(size_t)b*CAP + i];
  int ns = counts[b]; if (ns > K_MAX) ns = K_MAX;
  if (tid == 0) nsel[b] = ns;
  __syncthreads();
  for (int s = 0; s < K_MAX; s++) {
    if (s < ns) {
      unsigned long long mn = ~0ULL;
      for (int i = tid; i < cc; i += 256) { unsigned long long v = keys[i]; if (v < mn) mn = v; }
      red[tid] = mn;
      __syncthreads();
      for (int o = 128; o > 0; o >>= 1) {
        if (tid < o && red[tid+o] < red[tid]) red[tid] = red[tid+o];
        __syncthreads();
      }
      unsigned long long best = red[0];
      __syncthreads();
      for (int i = tid; i < cc; i += 256)
        if (keys[i] == best) { keys[i] = ~0ULL; sel[b*K_MAX + s] = (int)(unsigned)(best & 0xffffffffu); }
      __syncthreads();
    } else if (tid == 0) {
      sel[b*K_MAX + s] = 0;
    }
  }
}

// ---------------------------------------------------------------------------
// K3: normalized local coords, 4-NN density, token assembly.
// ---------------------------------------------------------------------------
__launch_bounds__(64)
__global__ void k_geom(const float* __restrict__ pts,
                       const float* __restrict__ boxes,
                       const int* __restrict__ sel,
                       const int* __restrict__ nsel,
                       float* __restrict__ tokens) {
  __shared__ float snl[K_MAX][3];
  int b = blockIdx.x, k = threadIdx.x;
  int ns = nsel[b];
  float cx = boxes[b*7+0], cy = boxes[b*7+1], cz = boxes[b*7+2];
  float dx = boxes[b*7+3], dy = boxes[b*7+4], dz = boxes[b*7+5];
  float yaw = boxes[b*7+6];
  float c = cosf(yaw), s = sinf(yaw);
  float ddx = fmaxf(dx, 0.001f), ddy = fmaxf(dy, 0.001f), ddz = fmaxf(dz, 0.001f);
  float nx = 0.f, ny = 0.f, nz = 0.f, f0 = 0.f, f1 = 0.f, f2 = 0.f, f3 = 0.f;
  if (k < ns) {
    int i = sel[b*K_MAX + k];
    float px = pts[i*7+0], py = pts[i*7+1], pz = pts[i*7+2];
    f0 = pts[i*7+3]; f1 = pts[i*7+4]; f2 = pts[i*7+5]; f3 = pts[i*7+6];
    float rx = px-cx, ry = py-cy, rz = pz-cz;
    float lx =  rx*c + ry*s;
    float ly = -rx*s + ry*c;
    nx = lx/ddx; ny = ly/ddy; nz = rz/ddz - 0.5f;
  }
  snl[k][0] = nx; snl[k][1] = ny; snl[k][2] = nz;
  __syncthreads();
  float best[4] = {1e30f, 1e30f, 1e30f, 1e30f};
  for (int j = 0; j < ns; j++) {
    float ax = nx - snl[j][0], ay = ny - snl[j][1], az = nz - snl[j][2];
    float d = sqrtf(ax*ax + ay*ay + az*az + 1e-12f);
    if (d < best[3]) {
      best[3] = d;
      #pragma unroll
      for (int t = 3; t > 0; t--)
        if (best[t] < best[t-1]) { float tmp = best[t]; best[t] = best[t-1]; best[t-1] = tmp; }
    }
  }
  int m4 = (ns < 4) ? ns : 4;
  float sum = 0.f;
  #pragma unroll
  for (int ci = 1; ci < 4; ci++) if (ci < m4) sum += best[ci];
  float nnb = fmaxf((float)(m4 - 1), 1.f);
  float dens = (k < ns) ? expf(-sum / nnb) : 0.f;
  float* t = tokens + ((size_t)b*K_MAX + k)*8;
  t[0] = nx; t[1] = ny; t[2] = nz; t[3] = f0; t[4] = f1; t[5] = f2; t[6] = f3; t[7] = dens;
}

// ---------------------------------------------------------------------------
// K4: layer 1 (K=8 -> 256, relu), emits f16 activations for the WMMA chain.
// ---------------------------------------------------------------------------
__launch_bounds__(256)
__global__ void k_layer1(const float* __restrict__ tokens,
                         const float* __restrict__ w1,
                         const float* __restrict__ b1,
                         _Float16* __restrict__ h1f16) {
  int m = blockIdx.x, n = threadIdx.x;
  __shared__ float t[8];
  if (n < 8) t[n] = tokens[(size_t)m*8 + n];
  __syncthreads();
  float acc = b1[n];
  #pragma unroll
  for (int c = 0; c < 8; c++) acc += t[c] * w1[c*HID + n];
  h1f16[(size_t)m*HID + n] = (_Float16)fmaxf(acc, 0.f);
}

// ---------------------------------------------------------------------------
// Weight prep: f32 [K][N] -> f16 transposed [N][K] (k-contiguous, so B tiles
// are straight 16B async copies).
// ---------------------------------------------------------------------------
__global__ void k_transpose_f16(const float* __restrict__ W, int K, int N,
                                _Float16* __restrict__ WT) {
  int idx = blockIdx.x * 256 + threadIdx.x;
  if (idx >= K * N) return;
  int n = idx / K, k = idx - n * K;
  WT[(size_t)n*K + k] = (_Float16)W[(size_t)k*N + n];
}

// W'f transposed: [256][128], k<96 -> f_w[k][n], k==96 -> f_b[n], else 0.
__global__ void k_prepwfT(const float* __restrict__ fw,
                          const float* __restrict__ fb,
                          _Float16* __restrict__ WfT) {
  int idx = blockIdx.x * 256 + threadIdx.x;   // 256*128
  int n = idx >> 7, k = idx & 127;
  float v = (k < C_IMG) ? fw[(size_t)k*HID + n] : ((k == C_IMG) ? fb[n] : 0.f);
  WfT[idx] = (_Float16)v;
}

// ---------------------------------------------------------------------------
// WMMA GEMM, f16 x f16 -> f32. Block tile 64x64, 8 waves each computing
// 16x32 (two v_wmma per K-step sharing one A fragment). Double-buffered LDS
// filled with global_load_async_to_lds_b128; s_wait_asynccnt pipelining.
//   C32[m,n] and/or C16[m,n] = rs[m] * act(A@W + bias[n]) + res[m,n]
// ---------------------------------------------------------------------------
__launch_bounds__(256)
__global__ void gemm_wmma(const _Float16* __restrict__ A, int lda,
                          const _Float16* __restrict__ WT, int ldw, // [N][K]
                          const float* __restrict__ bias,
                          const float* __restrict__ rs,
                          const float* __restrict__ res, int ldres,
                          float* __restrict__ C, int ldc,
                          _Float16* __restrict__ C16, int ldc16,
                          int K, int relu) {
  __shared__ __align__(16) _Float16 As[2][64][40];  // 80B rows = 5x16B
  __shared__ __align__(16) _Float16 Bs[2][64][40];
  int tid  = threadIdx.x;
  int lane = tid & 31;
  int wave = tid >> 5;
  int wm = wave >> 1;          // 0..3  : 16-row slab
  int wn = wave & 1;           // 0..1  : 32-col slab (2 subtiles)
  int hl = lane >> 4;
  int r  = lane & 15;
  int blockM = blockIdx.x * 64;
  int blockN = blockIdx.y * 64;

  int srow = tid >> 2;                 // 0..63
  int sc8  = (tid & 3) * 8;            // 0,8,16,24
  const _Float16* agp = A  + (size_t)(blockM + srow)*lda + sc8;
  const _Float16* bgp = WT + (size_t)(blockN + srow)*ldw + sc8;

  v8f acc0, acc1;
  #pragma unroll
  for (int i = 0; i < 8; i++) { acc0[i] = 0.f; acc1[i] = 0.f; }

  // stage tile 0
  async_copy_b128((unsigned)(size_t)&As[0][srow][sc8], agp);
  async_copy_b128((unsigned)(size_t)&Bs[0][srow][sc8], bgp);

  int nk = K >> 5;
  for (int t = 0; t < nk; t++) {
    int cur = t & 1;
    if (t + 1 < nk) {
      int nxt = cur ^ 1;
      int k1 = (t + 1) << 5;
      async_copy_b128((unsigned)(size_t)&As[nxt][srow][sc8], agp + k1);
      async_copy_b128((unsigned)(size_t)&Bs[nxt][srow][sc8], bgp + k1);
      wait_async_le2();                // tile t's copies have landed
    } else {
      wait_async_0();
    }
    __syncthreads();                   // all waves' tile-t data visible

    v16h a, b0, b1;
    int am  = wm*16 + r;
    int bn0 = wn*32 + r;
    int bn1 = wn*32 + 16 + r;
    #pragma unroll
    for (int v = 0; v < 8; v++) {
      int ka = ((v >> 2) << 4) + hl*8 + ((v & 3) << 1);  // A: ISA 16x32 f16 layout
      int kb = ka;                                        // B: same K order as A
      a[2*v]    = As[cur][am][ka];
      a[2*v+1]  = As[cur][am][ka+1];
      b0[2*v]   = Bs[cur][bn0][kb];
      b0[2*v+1] = Bs[cur][bn0][kb+1];
      b1[2*v]   = Bs[cur][bn1][kb];
      b1[2*v+1] = Bs[cur][bn1][kb+1];
    }
    acc0 = __builtin_amdgcn_wmma_f32_16x16x32_f16(false, a, false, b0,
                                                  (short)0, acc0, false, false);
    acc1 = __builtin_amdgcn_wmma_f32_16x16x32_f16(false, a, false, b1,
                                                  (short)0, acc1, false, false);
    __syncthreads();                   // done reading buf before it is refilled
  }

  #pragma unroll
  for (int j = 0; j < 2; j++) {
    const v8f& acc = j ? acc1 : acc0;
    int n = blockN + wn*32 + j*16 + r;
    float bv = bias ? bias[n] : 0.f;
    #pragma unroll
    for (int i = 0; i < 8; i++) {
      int m = blockM + wm*16 + hl*8 + i;
      float v = acc[i] + bv;
      if (relu) v = fmaxf(v, 0.f);
      if (rs)  v *= rs[m];
      if (res) v += res[(size_t)m*ldres + n];
      if (C)   C[(size_t)m*ldc + n] = v;
      if (C16) C16[(size_t)m*ldc16 + n] = (_Float16)v;
    }
  }
}

// ---------------------------------------------------------------------------
// K6: deformable sampling. One wave per (box,slot); bilinear sampling commutes
// with the 1x1 conv f_w, so gather raw 96-ch image + bias weight scalar.
// Emits f16 ctx rows (padded to 128) feeding the context WMMA GEMM.
// ---------------------------------------------------------------------------
__launch_bounds__(256)
__global__ void k_sample(const float* __restrict__ h,
                         const float* __restrict__ pts,
                         const int* __restrict__ sel,
                         const int* __restrict__ nsel,
                         const float* __restrict__ proj,
                         const float* __restrict__ img,
                         const float* __restrict__ off_w,
                         const float* __restrict__ off_b,
                         const float* __restrict__ wt_w,
                         const float* __restrict__ wt_b,
                         _Float16* __restrict__ ctx,
                         float* __restrict__ vm) {
  int wave = threadIdx.x >> 5, lane = threadIdx.x & 31;
  int slot = blockIdx.x * 8 + wave;
  int b = slot >> 6, k = slot & 63;
  int ns = nsel[b];
  bool valid_slot = (k < ns);
  _Float16* crow = ctx + (size_t)slot * 128;

  float p[12];
  #pragma unroll
  for (int j = 0; j < 12; j++) p[j] = 0.f;
  const float* hr = h + (size_t)slot * HID;
  #pragma unroll
  for (int i = 0; i < 8; i++) {
    int ch = lane*8 + i;
    float hv = hr[ch];
    #pragma unroll
    for (int j = 0; j < 8; j++) p[j]   += hv * off_w[ch*8 + j];
    #pragma unroll
    for (int j = 0; j < 4; j++) p[8+j] += hv * wt_w[ch*4 + j];
  }
  #pragma unroll
  for (int j = 0; j < 12; j++)
    for (int o = 16; o > 0; o >>= 1) p[j] += __shfl_xor(p[j], o, 32);

  float ox[SAMP], oy[SAMP];
  #pragma unroll
  for (int s2 = 0; s2 < SAMP; s2++) {
    ox[s2] = tanhf(p[2*s2]   + off_b[2*s2])   * 0.04f;
    oy[s2] = tanhf(p[2*s2+1] + off_b[2*s2+1]) * 0.04f;
  }
  float wl[SAMP], mx = -1e30f;
  #pragma unroll
  for (int s2 = 0; s2 < SAMP; s2++) { wl[s2] = p[8+s2] + wt_b[s2]; mx = fmaxf(mx, wl[s2]); }
  float ssum = 0.f;
  #pragma unroll
  for (int s2 = 0; s2 < SAMP; s2++) { wl[s2] = expf(wl[s2] - mx); ssum += wl[s2]; }

  float vmf = 0.f, rx = 0.f, ry = 0.f;
  if (valid_slot) {
    int i = sel[b*K_MAX + k];
    float X = pts[i*7+0], Y = pts[i*7+1], Z = pts[i*7+2];
    float h0  = X*proj[0] + Y*proj[1] + Z*proj[2]  + proj[3];
    float h1v = X*proj[4] + Y*proj[5] + Z*proj[6]  + proj[7];
    float dep = X*proj[8] + Y*proj[9] + Z*proj[10] + proj[11];
    float sd = (fabsf(dep) > 0.001f) ? dep : 0.001f;
    rx = 2.f*(h0  / sd) / (IMG_W_F - 1.f) - 1.f;
    ry = 2.f*(h1v / sd) / (IMG_H_F - 1.f) - 1.f;
    if (dep > 0.1f && fabsf(rx) <= 1.f && fabsf(ry) <= 1.f) vmf = 1.f;
  }

  float wt4[SAMP], wsum = 0.f;
  #pragma unroll
  for (int s2 = 0; s2 < SAMP; s2++) { wt4[s2] = (wl[s2]/ssum) * vmf; wsum += wt4[s2]; }
  float inv = 1.f / fmaxf(wsum, 1e-6f);
  #pragma unroll
  for (int s2 = 0; s2 < SAMP; s2++) wt4[s2] *= inv;

  float c0 = 0.f, c1 = 0.f, c2 = 0.f, bs = 0.f;
  if (vmf > 0.f) {
    #pragma unroll
    for (int s2 = 0; s2 < SAMP; s2++) {
      float gx = fminf(fmaxf(rx + ox[s2], -1.2f), 1.2f);
      float gy = fminf(fmaxf(ry + oy[s2], -1.2f), 1.2f);
      float pxf = (gx + 1.f) * 0.5f * (WF - 1);
      float pyf = (gy + 1.f) * 0.5f * (HF - 1);
      float x0 = floorf(pxf), y0 = floorf(pyf);
      float wx = pxf - x0, wy = pyf - y0;
      float ws4 = wt4[s2];
      #pragma unroll
      for (int cn = 0; cn < 4; cn++) {
        float xi = x0 + (float)(cn & 1), yi = y0 + (float)(cn >> 1);
        float wc = ((cn & 1) ? wx : 1.f - wx) * ((cn >> 1) ? wy : 1.f - wy);
        if (xi >= 0.f && xi <= (float)(WF-1) && yi >= 0.f && yi <= (float)(HF-1)) {
          int xc = (int)xi, yc = (int)yi;
          float wcs = ws4 * wc;
          const float* base = img + (size_t)yc * WF + xc;
          c0 += wcs * base[(size_t)(lane)      * HF * WF];
          c1 += wcs * base[(size_t)(lane + 32) * HF * WF];
          c2 += wcs * base[(size_t)(lane + 64) * HF * WF];
          bs += wcs;
        }
      }
    }
  }
  crow[lane]      = (_Float16)c0;
  crow[lane + 32] = (_Float16)c1;
  crow[lane + 64] = (_Float16)c2;
  crow[96 + lane] = (lane == 0) ? (_Float16)bs : (_Float16)0.f;
  if (lane == 0) vm[slot] = vmf;
}

// ---------------------------------------------------------------------------
// K7: score head, per-box softmax over slots, pooled output + counts.
// ---------------------------------------------------------------------------
__launch_bounds__(256)
__global__ void k_pool(const float* __restrict__ fused,
                       const float* __restrict__ score_w,
                       const float* __restrict__ score_b,
                       const int* __restrict__ nsel,
                       const int* __restrict__ counts,
                       float* __restrict__ out) {
  __shared__ float lg[K_MAX];
  __shared__ float pw[K_MAX];
  int b = blockIdx.x, tid = threadIdx.x;
  int ns = nsel[b];
  if (tid < K_MAX) {
    float acc = score_b[0];
    const float* fr = fused + ((size_t)b*K_MAX + tid) * HID;
    for (int c = 0; c < HID; c++) acc += fr[c] * score_w[c];
    lg[tid] = acc;
  }
  __syncthreads();
  if (tid == 0) {
    if (ns > 0) {
      float mx = -1e30f;
      for (int k = 0; k < ns; k++) mx = fmaxf(mx, lg[k]);
      float ssum = 0.f;
      for (int k = 0; k < K_MAX; k++) {
        float e = (k < ns) ? expf(lg[k] - mx) : 0.f;
        pw[k] = e; ssum += e;
      }
      float inv = 1.f / ssum;
      for (int k = 0; k < K_MAX; k++) pw[k] *= inv;
    } else {
      for (int k = 0; k < K_MAX; k++) pw[k] = 0.f;
    }
  }
  __syncthreads();
  float acc = 0.f;
  for (int k = 0; k < K_MAX; k++)
    acc += pw[k] * fused[((size_t)b*K_MAX + k) * HID + tid];
  out[(size_t)b*HID + tid] = acc;
  if (tid == 0) out[(size_t)P_BOX*HID + b] = (float)counts[b];
}

// ---------------------------------------------------------------------------
extern "C" void kernel_launch(void* const* d_in, const int* in_sizes, int n_in,
                              void* d_out, int out_size, void* d_ws, size_t ws_size,
                              hipStream_t stream) {
  (void)in_sizes; (void)n_in; (void)out_size; (void)ws_size;
  const float* points  = (const float*)d_in[0];
  const float* boxes   = (const float*)d_in[1];
  const float* image   = (const float*)d_in[2];
  const float* proj    = (const float*)d_in[3];
  const float* pe_w1   = (const float*)d_in[4];
  const float* pe_b1   = (const float*)d_in[5];
  const float* pe_w2   = (const float*)d_in[6];
  const float* pe_b2   = (const float*)d_in[7];
  const float* q_w     = (const float*)d_in[8];
  const float* q_b     = (const float*)d_in[9];
  const float* f_w     = (const float*)d_in[10];
  const float* f_b     = (const float*)d_in[11];
  const float* off_w   = (const float*)d_in[12];
  const float* off_b   = (const float*)d_in[13];
  const float* wt_w    = (const float*)d_in[14];
  const float* wt_b    = (const float*)d_in[15];
  const float* o_w1    = (const float*)d_in[16];
  const float* o_b1    = (const float*)d_in[17];
  const float* o_w2    = (const float*)d_in[18];
  const float* o_b2    = (const float*)d_in[19];
  const float* score_w = (const float*)d_in[20];
  const float* score_b = (const float*)d_in[21];
  float* out = (float*)d_out;

  char* wsb = (char*)d_ws;
  size_t off = 0;
  auto take = [&](size_t bytes) -> char* {
    char* p = wsb + off;
    off = (off + bytes + 255) & ~(size_t)255;
    return p;
  };
  unsigned long long* cand = (unsigned long long*)take((size_t)P_BOX * CAP * 8);
  int*   counts  = (int*)take(P_BOX * 4);
  int*   candcnt = (int*)take(P_BOX * 4);
  int*   nselb   = (int*)take(P_BOX * 4);
  int*   selb    = (int*)take((size_t)P_BOX * K_MAX * 4);
  float* tokens  = (float*)take((size_t)M_TOT * 8 * 4);
  float* hbuf    = (float*)take((size_t)M_TOT * HID * 4);   // h (f32)
  float* fused   = (float*)take((size_t)M_TOT * HID * 4);
  float* vmb     = (float*)take((size_t)M_TOT * 4);
  _Float16* h1f   = (_Float16*)take((size_t)M_TOT * HID * 2);
  _Float16* h16   = (_Float16*)take((size_t)M_TOT * HID * 2);
  _Float16* ctx16 = (_Float16*)take((size_t)M_TOT * 128 * 2);
  _Float16* cat16 = (_Float16*)take((size_t)M_TOT * 2 * HID * 2);
  _Float16* o116  = (_Float16*)take((size_t)M_TOT * HID * 2);
  _Float16* pe_w2T = (_Float16*)take((size_t)HID * HID * 2);
  _Float16* q_wT   = (_Float16*)take((size_t)HID * HID * 2);
  _Float16* o_w1T  = (_Float16*)take((size_t)HID * 2 * HID * 2);
  _Float16* o_w2T  = (_Float16*)take((size_t)HID * HID * 2);
  _Float16* WfT    = (_Float16*)take((size_t)HID * 128 * 2);

  // weight prep (transpose + f16)
  k_transpose_f16<<<(HID*HID)/256, 256, 0, stream>>>(pe_w2, HID, HID, pe_w2T);
  k_transpose_f16<<<(HID*HID)/256, 256, 0, stream>>>(q_w, HID, HID, q_wT);
  k_transpose_f16<<<(2*HID*HID)/256, 256, 0, stream>>>(o_w1, 2*HID, HID, o_w1T);
  k_transpose_f16<<<(HID*HID)/256, 256, 0, stream>>>(o_w2, HID, HID, o_w2T);
  k_prepwfT<<<(HID*128)/256, 256, 0, stream>>>(f_w, f_b, WfT);

  // 1) candidate compaction + counts
  k_compact<<<P_BOX, 256, 0, stream>>>(points, boxes, cand, counts, candcnt);
  // 2) select 64 smallest-planar inside points per box
  k_select<<<P_BOX, 256, 0, stream>>>(cand, counts, candcnt, selb, nselb);
  // 3) geometry + density + tokens
  k_geom<<<P_BOX, 64, 0, stream>>>(points, boxes, selb, nselb, tokens);
  // 4) layer 1 -> f16
  k_layer1<<<M_TOT, 256, 0, stream>>>(tokens, pe_w1, pe_b1, h1f);

  dim3 gg(M_TOT / 64, HID / 64);
  // 5) h = h1 @ pe_w2 + pe_b2  (f32 for heads/residual, f16 for next GEMM)
  gemm_wmma<<<gg, 256, 0, stream>>>(h1f, HID, pe_w2T, HID, pe_b2,
                                    nullptr, nullptr, 0,
                                    hbuf, HID, h16, HID, HID, 0);
  // 6) deformable sampling -> ctx97 (f16, padded 128) + vm
  k_sample<<<M_TOT / 8, 256, 0, stream>>>(hbuf, points, selb, nselb, proj, image,
                                          off_w, off_b, wt_w, wt_b, ctx16, vmb);
  // 7) q * vm -> cat16[:, 0:256]
  gemm_wmma<<<gg, 256, 0, stream>>>(h16, HID, q_wT, HID, q_b,
                                    vmb, nullptr, 0,
                                    nullptr, 0, cat16, 2*HID, HID, 0);
  // 8) context = vm * (ctx97 @ W'f) -> cat16[:, 256:512]
  gemm_wmma<<<gg, 256, 0, stream>>>(ctx16, 128, WfT, 128, nullptr,
                                    vmb, nullptr, 0,
                                    nullptr, 0, cat16 + HID, 2*HID, 128, 0);
  // 9) o1 = relu(cat @ o_w1 + o_b1) -> f16
  gemm_wmma<<<gg, 256, 0, stream>>>(cat16, 2*HID, o_w1T, 2*HID, o_b1,
                                    nullptr, nullptr, 0,
                                    nullptr, 0, o116, HID, 2*HID, 1);
  // 10) fused = h + (o1 @ o_w2 + o_b2)  (f32)
  gemm_wmma<<<gg, 256, 0, stream>>>(o116, HID, o_w2T, HID, o_b2,
                                    nullptr, hbuf, HID,
                                    fused, HID, nullptr, 0, HID, 0);
  // 11) score + per-box softmax pooling + counts output
  k_pool<<<P_BOX, 256, 0, stream>>>(fused, score_w, score_b, nselb, counts, out);
}
